// LatentToProfileAttentionDecoder_46153718563371
// MI455X (gfx1250) — matrix operand order repaired and checked by reading, not compiled, not run on hardware
//
#include <hip/hip_runtime.h>
#include <hip/hip_bf16.h>

typedef __attribute__((ext_vector_type(16))) _Float16 v16h;
typedef __attribute__((ext_vector_type(8)))  _Float16 v8h;
typedef __attribute__((ext_vector_type(8)))  float    v8f;

#define T_STEPS 31
#define BATCH   65536

// ---- packed f16 weight fragment layout (B-matrix fragments, 512 f16 each) ----
#define NF_ATTN (4*6)     // N=64 (4 tiles) x K=192 (6 ksteps)
#define NF_CTX  (8*2)     // N=128 x K=64
#define NF_IN   (8*5)     // N=128 x K=160 (144 padded)
#define NF_GATE (32*8)    // N=512 x K=256 ([x;h] @ [W_ih;W_hh])
#define OFF_ATTN 0
#define OFF_CTX  (OFF_ATTN + NF_ATTN*512)
#define OFF_IN   (OFF_CTX  + NF_CTX *512)
#define OFF_GATE (OFF_IN   + NF_IN  *512)
#define PW_ELEMS (OFF_GATE + NF_GATE*512)   // 172032 f16
#define PW_BYTES (PW_ELEMS*2)               // 344064 bytes (16B aligned)

// ---- packed f32 bias/vector region (after PW) ----
#define B_ATTN 0      // 64
#define B_CTX  64     // 128
#define B_IN   192    // 128
#define B_GATE 320    // 512  (b_ih + b_hh)
#define B_OUTW 832    // 128
#define B_OUTB 960    // 1
#define NB     961

// =====================================================================
// Prep: pack weights into WMMA B fragments (f16, contiguous k order) + biases
// B 32x16 f16 packing (this code's contiguous k order, not AMD's table): lane l -> column n = tile_n*16 + (l&15);
//   lanes 0-15 hold K = ktile*32 + 0..15, lanes 16-31 hold K = +16..31,
//   16 f16 per lane stored contiguously.
// =====================================================================
__global__ void pack_weights_kernel(
    const float* __restrict__ attn_W, const float* __restrict__ ctx_W,
    const float* __restrict__ in_W,   const float* __restrict__ W_ih,
    const float* __restrict__ W_hh,   const float* __restrict__ attn_b,
    const float* __restrict__ ctx_b,  const float* __restrict__ in_b,
    const float* __restrict__ b_ih,   const float* __restrict__ b_hh,
    const float* __restrict__ out_W,  const float* __restrict__ out_b,
    _Float16* __restrict__ pW, float* __restrict__ pB)
{
    int e = blockIdx.x * blockDim.x + threadIdx.x;
    if (e < PW_ELEMS) {
        int region, base, numK, Kreal, Kw;
        if      (e < OFF_CTX)  { region = 0; base = OFF_ATTN; numK = 6; Kreal = 192; Kw = 192; }
        else if (e < OFF_IN)   { region = 1; base = OFF_CTX;  numK = 2; Kreal = 64;  Kw = 64;  }
        else if (e < OFF_GATE) { region = 2; base = OFF_IN;   numK = 5; Kreal = 144; Kw = 144; }
        else                   { region = 3; base = OFF_GATE; numK = 8; Kreal = 256; Kw = 128; }
        int e2   = e - base;
        int f    = e2 >> 9;
        int w512 = e2 & 511;
        int lane = w512 >> 4;
        int j    = w512 & 15;
        int ktile = f % numK, ntile = f / numK;
        int n = ntile * 16 + (lane & 15);
        int k = ktile * 32 + ((lane < 16) ? 0 : 16) + j;
        float val = 0.0f;
        if (k < Kreal) {
            if      (region == 0) val = attn_W[n * Kw + k];
            else if (region == 1) val = ctx_W [n * Kw + k];
            else if (region == 2) val = in_W  [n * Kw + k];
            else                  val = (k < 128) ? W_ih[n * 128 + k]
                                                  : W_hh[n * 128 + (k - 128)];
        }
        pW[e] = (_Float16)val;
    } else {
        int b = e - PW_ELEMS;
        if (b < NB) {
            float v;
            if      (b < 64)   v = attn_b[b];
            else if (b < 192)  v = ctx_b[b - 64];
            else if (b < 320)  v = in_b[b - 192];
            else if (b < 832)  v = b_ih[b - 320] + b_hh[b - 320];
            else if (b < 960)  v = out_W[b - 832];
            else               v = out_b[0];
            pB[b] = v;
        }
    }
}

// =====================================================================
// WMMA helpers
// =====================================================================
__device__ __forceinline__ v16h load_a_frag(const _Float16* row0, int lane, int ldk) {
    // A 16x32, contiguous k order (not AMD's table order): lane m=l&15; lanes 0-15 K=0..15, lanes 16-31 K=16..31 (matches packed B)
    const _Float16* p = row0 + (lane & 15) * ldk + ((lane < 16) ? 0 : 16);
    v8h lo = *(const v8h*)(p);
    v8h hi = *(const v8h*)(p + 8);
    v16h r;
#pragma unroll
    for (int i = 0; i < 8; ++i) { r[i] = lo[i]; r[8 + i] = hi[i]; }
    return r;
}

__device__ __forceinline__ v16h load_b_frag(const _Float16* pw, int frag, int lane) {
    return *(const v16h*)(pw + frag * 512 + lane * 16);  // 32B contiguous per lane
}

__device__ __forceinline__ v8f wmma_f16(v16h a, v16h b, v8f c) {
    return __builtin_amdgcn_wmma_f32_16x16x32_f16(false, a, false, b, (short)0, c, false, false);
}

__device__ __forceinline__ float sigmoidf_(float x) { return 1.0f / (1.0f + __expf(-x)); }
__device__ __forceinline__ float tanhf_(float x)    { return 1.0f - 2.0f / (1.0f + __expf(2.0f * x)); }

// =====================================================================
// Main fused recurrent decoder.
// 2 waves/block; each wave owns 32 batch rows = two 16-row M-tiles, so every
// B fragment loaded from L2 feeds two WMMAs (halves weight-stream traffic).
// State: h in LDS f16 (double-buffered across gate phase), c in LDS f32.
// =====================================================================
__global__ __launch_bounds__(64) void decoder_kernel(
    const float* __restrict__ z, const float* __restrict__ emb,
    const _Float16* __restrict__ pW, const float* __restrict__ pB,
    float* __restrict__ out)
{
    __shared__ __attribute__((aligned(16))) _Float16 sHZ[2][32][192];  // [h(128) | z(64)] f16
    __shared__ __attribute__((aligned(16))) _Float16 sHN[2][32][128];  // h_new staging
    __shared__ __attribute__((aligned(16))) _Float16 sAZ[2][32][64];   // attn*z
    __shared__ __attribute__((aligned(16))) _Float16 sXC[2][32][160];  // [de(16)|ctx(128)|pad(16)]
    __shared__ __attribute__((aligned(16))) float    sC [2][32][128];  // cell state f32
    __shared__ __attribute__((aligned(16))) unsigned char uSLX[2][8192]; // sLog f32[32][64] / sX f16[32][128]
    __shared__ __attribute__((aligned(16))) float    sB[NB];

    const int tid  = threadIdx.x;
    const int w    = tid >> 5;
    const int lane = tid & 31;
    const int tileBase = (blockIdx.x * 2 + w) * 32;
    const int mlo  = lane & 15;
    const int mAdd = (lane >> 4) * 8;        // D/C fragment row base per lane half

    float*    sLogW = (float*)uSLX[w];       // [32][64] f32 (attn logits)
    _Float16* sXW   = (_Float16*)uSLX[w];    // [32][128] f16 (relu proj), disjoint live range

    const _Float16* pAttn = pW + OFF_ATTN;
    const _Float16* pCtx  = pW + OFF_CTX;
    const _Float16* pIn   = pW + OFF_IN;
    const _Float16* pGate = pW + OFF_GATE;

    for (int i = tid; i < NB; i += 64) sB[i] = pB[i];

    // init: h = 0, c = 0, z tile -> f16 LDS, zero in-proj K padding
    for (int i = lane; i < 32 * 128; i += 32) sHZ[w][i >> 7][i & 127] = (_Float16)0.0f;
    for (int i = lane; i < 32 * 128; i += 32) sC [w][i >> 7][i & 127] = 0.0f;
    for (int i = lane; i < 32 * 64; i += 32) {
        int m = i >> 6, n = i & 63;
        sHZ[w][m][128 + n] = (_Float16)z[(tileBase + m) * 64 + n];
    }
    for (int i = lane; i < 32 * 16; i += 32) sXC[w][i >> 4][144 + (i & 15)] = (_Float16)0.0f;
    __syncthreads();

#pragma unroll 1
    for (int t = 0; t < T_STEPS; ++t) {
        // depth embedding row broadcast into x-proj input
        for (int i = lane; i < 32 * 16; i += 32)
            sXC[w][i >> 4][i & 15] = (_Float16)emb[t * 16 + (i & 15)];

        // ---- attention logits: [32,192] @ [192,64] ----
#pragma unroll 1
        for (int nt = 0; nt < 4; ++nt) {
            v8f acc0 = {}, acc1 = {};
#pragma unroll
            for (int ks = 0; ks < 6; ++ks) {
                v16h b = load_b_frag(pAttn, nt * 6 + ks, lane);
                acc0 = wmma_f16(load_a_frag(&sHZ[w][ 0][ks * 32], lane, 192), b, acc0);
                acc1 = wmma_f16(load_a_frag(&sHZ[w][16][ks * 32], lane, 192), b, acc1);
            }
            int n = nt * 16 + mlo;
            float bb = sB[B_ATTN + n];
#pragma unroll
            for (int v = 0; v < 8; ++v) {
                sLogW[(v + mAdd) * 64 + n]      = acc0[v] + bb;
                sLogW[(16 + v + mAdd) * 64 + n] = acc1[v] + bb;
            }
        }
        __syncthreads();

        // ---- softmax over 64 latents, then attn*z (f16); one row per lane ----
        {
            const int r = lane;
            float mx = -3.4e38f;
            for (int n = 0; n < 64; ++n) mx = fmaxf(mx, sLogW[r * 64 + n]);
            float s = 0.0f;
            for (int n = 0; n < 64; ++n) s += __expf(sLogW[r * 64 + n] - mx);
            float inv = 1.0f / s;
            for (int n = 0; n < 64; ++n) {
                float p  = __expf(sLogW[r * 64 + n] - mx) * inv;
                float zf = (float)sHZ[w][r][128 + n];
                sAZ[w][r][n] = (_Float16)(p * zf);
            }
        }
        __syncthreads();

        // ---- context: [32,64] @ [64,128] ----
#pragma unroll 1
        for (int nt = 0; nt < 8; ++nt) {
            v8f acc0 = {}, acc1 = {};
#pragma unroll
            for (int ks = 0; ks < 2; ++ks) {
                v16h b = load_b_frag(pCtx, nt * 2 + ks, lane);
                acc0 = wmma_f16(load_a_frag(&sAZ[w][ 0][ks * 32], lane, 64), b, acc0);
                acc1 = wmma_f16(load_a_frag(&sAZ[w][16][ks * 32], lane, 64), b, acc1);
            }
            int n = nt * 16 + mlo;
            float bb = sB[B_CTX + n];
#pragma unroll
            for (int v = 0; v < 8; ++v) {
                sXC[w][v + mAdd][16 + n]      = (_Float16)(acc0[v] + bb);
                sXC[w][16 + v + mAdd][16 + n] = (_Float16)(acc1[v] + bb);
            }
        }
        __syncthreads();

        // ---- input projection + ReLU: [32,160] @ [160,128] ----
#pragma unroll 1
        for (int nt = 0; nt < 8; ++nt) {
            v8f acc0 = {}, acc1 = {};
#pragma unroll
            for (int ks = 0; ks < 5; ++ks) {
                v16h b = load_b_frag(pIn, nt * 5 + ks, lane);
                acc0 = wmma_f16(load_a_frag(&sXC[w][ 0][ks * 32], lane, 160), b, acc0);
                acc1 = wmma_f16(load_a_frag(&sXC[w][16][ks * 32], lane, 160), b, acc1);
            }
            int n = nt * 16 + mlo;
            float bb = sB[B_IN + n];
#pragma unroll
            for (int v = 0; v < 8; ++v) {
                float x0 = acc0[v] + bb, x1 = acc1[v] + bb;
                sXW[(v + mAdd) * 128 + n]      = (_Float16)(x0 > 0.0f ? x0 : 0.0f);
                sXW[(16 + v + mAdd) * 128 + n] = (_Float16)(x1 > 0.0f ? x1 : 0.0f);
            }
        }
        __syncthreads();

        // ---- LSTM gates: [32,256]([x;h]) @ [256,512], fused cell update ----
        // reads old h from sHZ, writes new h to sHN (no in-place hazard)
#pragma unroll 1
        for (int ng = 0; ng < 8; ++ng) {
            v8f aI0 = {}, aF0 = {}, aG0 = {}, aO0 = {};
            v8f aI1 = {}, aF1 = {}, aG1 = {}, aO1 = {};
#pragma unroll
            for (int ks = 0; ks < 8; ++ks) {
                v16h a0 = (ks < 4) ? load_a_frag(sXW + ks * 32, lane, 128)
                                   : load_a_frag(&sHZ[w][ 0][(ks - 4) * 32], lane, 192);
                v16h a1 = (ks < 4) ? load_a_frag(sXW + 16 * 128 + ks * 32, lane, 128)
                                   : load_a_frag(&sHZ[w][16][(ks - 4) * 32], lane, 192);
                v16h bI = load_b_frag(pGate, ( 0 + ng) * 8 + ks, lane);
                aI0 = wmma_f16(a0, bI, aI0);  aI1 = wmma_f16(a1, bI, aI1);
                v16h bF = load_b_frag(pGate, ( 8 + ng) * 8 + ks, lane);
                aF0 = wmma_f16(a0, bF, aF0);  aF1 = wmma_f16(a1, bF, aF1);
                v16h bG = load_b_frag(pGate, (16 + ng) * 8 + ks, lane);
                aG0 = wmma_f16(a0, bG, aG0);  aG1 = wmma_f16(a1, bG, aG1);
                v16h bO = load_b_frag(pGate, (24 + ng) * 8 + ks, lane);
                aO0 = wmma_f16(a0, bO, aO0);  aO1 = wmma_f16(a1, bO, aO1);
            }
            int j = ng * 16 + mlo;
            float bi = sB[B_GATE + j],       bf = sB[B_GATE + 128 + j];
            float bg = sB[B_GATE + 256 + j], bo = sB[B_GATE + 384 + j];
#pragma unroll
            for (int v = 0; v < 8; ++v) {
                int m0 = v + mAdd, m1 = 16 + v + mAdd;
                float ig = sigmoidf_(aI0[v] + bi);
                float fg = sigmoidf_(aF0[v] + bf);
                float gg = tanhf_(aG0[v] + bg);
                float og = sigmoidf_(aO0[v] + bo);
                float cn = fg * sC[w][m0][j] + ig * gg;
                sC[w][m0][j] = cn;
                sHN[w][m0][j] = (_Float16)(og * tanhf_(cn));
                ig = sigmoidf_(aI1[v] + bi);
                fg = sigmoidf_(aF1[v] + bf);
                gg = tanhf_(aG1[v] + bg);
                og = sigmoidf_(aO1[v] + bo);
                cn = fg * sC[w][m1][j] + ig * gg;
                sC[w][m1][j] = cn;
                sHN[w][m1][j] = (_Float16)(og * tanhf_(cn));
            }
        }
        __syncthreads();

        // ---- publish h_new into [h|z] buffer for next step ----
        for (int i = lane; i < 32 * 128 / 8; i += 32) {   // 8 f16 per copy
            int m = (i * 8) >> 7, n = (i * 8) & 127;
            *(v8h*)&sHZ[w][m][n] = *(const v8h*)&sHN[w][m][n];
        }

        // ---- output projection v = h_new @ out_W.T + out_b; one row per lane ----
        {
            float vv = sB[B_OUTB];
            for (int jj = 0; jj < 128; ++jj)
                vv += (float)sHN[w][lane][jj] * sB[B_OUTW + jj];
            out[(tileBase + lane) * T_STEPS + t] = vv;
        }
        __syncthreads();
    }
}

// =====================================================================
extern "C" void kernel_launch(void* const* d_in, const int* in_sizes, int n_in,
                              void* d_out, int out_size, void* d_ws, size_t ws_size,
                              hipStream_t stream) {
    (void)in_sizes; (void)n_in; (void)out_size; (void)ws_size;
    const float* z      = (const float*)d_in[0];
    const float* emb    = (const float*)d_in[1];
    const float* attn_W = (const float*)d_in[2];
    const float* attn_b = (const float*)d_in[3];
    const float* ctx_W  = (const float*)d_in[4];
    const float* ctx_b  = (const float*)d_in[5];
    const float* in_W   = (const float*)d_in[6];
    const float* in_b   = (const float*)d_in[7];
    const float* W_ih   = (const float*)d_in[8];
    const float* W_hh   = (const float*)d_in[9];
    const float* b_ih   = (const float*)d_in[10];
    const float* b_hh   = (const float*)d_in[11];
    const float* out_W  = (const float*)d_in[12];
    const float* out_b  = (const float*)d_in[13];

    _Float16* pW = (_Float16*)d_ws;
    float*    pB = (float*)((char*)d_ws + PW_BYTES);

    int packTotal = PW_ELEMS + NB;
    pack_weights_kernel<<<(packTotal + 255) / 256, 256, 0, stream>>>(
        attn_W, ctx_W, in_W, W_ih, W_hh,
        attn_b, ctx_b, in_b, b_ih, b_hh, out_W, out_b, pW, pB);

    decoder_kernel<<<BATCH / 64, 64, 0, stream>>>(
        z, emb, pW, pB, (float*)d_out);
}
